// Model_76957224009934
// MI455X (gfx1250) — compile-verified, matrix operand order checked by reading, not run on hardware
//
#include <hip/hip_runtime.h>
#include <hip/hip_bf16.h>
#include <math.h>

typedef __attribute__((ext_vector_type(16))) _Float16 v16h;
typedef __attribute__((ext_vector_type(8)))  float    v8f;

#define T_LEN 262144
#define HDIM  48
#define GATES 192              // 4*H
#define MTILES (T_LEN / 16)    // 16384 M-tiles of 16 timesteps

// ---------------------------------------------------------------------------
// Phase A: P0[t][j] = sum_k x[t][k]*W_ih0[j][k] + b_ih0[j] + b_hh0[j]
// GEMM [T x 5] * [5 x 192] done with v_wmma_f32_16x16x32_f16 (K padded 5->32).
// One wave handles one 16-timestep M-tile and loops the 12 N-tiles of 16 gates.
// ---------------------------------------------------------------------------
__global__ __launch_bounds__(256) void lstm_inproj_wmma(
    const float* __restrict__ x, const float* __restrict__ Wih0,
    const float* __restrict__ bih0, const float* __restrict__ bhh0,
    float* __restrict__ P0)
{
  const int lane = threadIdx.x & 31;
  const int wid  = threadIdx.x >> 5;
  const int mt   = blockIdx.x * 8 + wid;      // grid sized so mt < MTILES always
  const int t0   = mt << 4;

  // A fragment: 16x32 f16. Per ISA layout, lanes 0-15 elements 0..7 hold K=0..7
  // for row M=lane; everything with K>=5 (and all of lanes 16-31) is zero.
  v16h a = {};
  if (lane < 16) {
    const float* xr = x + (size_t)(t0 + lane) * 5;
    a[0] = (_Float16)xr[0]; a[1] = (_Float16)xr[1]; a[2] = (_Float16)xr[2];
    a[3] = (_Float16)xr[3]; a[4] = (_Float16)xr[4];
  }

  for (int n = 0; n < 12; ++n) {
    // B fragment: 32x16 f16, element (K,N)=W_ih0[n*16+N][K]; lanes 0-15 hold
    // K=0..15 (element e -> K=e), lanes 16-31 hold K=16..31 (all zero here).
    v16h b = {};
    if (lane < 16) {
      const float* wr = Wih0 + (size_t)(n * 16 + lane) * 5;
      b[0] = (_Float16)wr[0]; b[1] = (_Float16)wr[1]; b[2] = (_Float16)wr[2];
      b[3] = (_Float16)wr[3]; b[4] = (_Float16)wr[4];
    }
    const int   col = n * 16 + (lane & 15);
    const float bs  = bih0[col] + bhh0[col];   // bias depends only on N
    v8f c;
    #pragma unroll
    for (int r = 0; r < 8; ++r) c[r] = bs;

    v8f d = __builtin_amdgcn_wmma_f32_16x16x32_f16(
        /*neg_a=*/false, a, /*neg_b=*/false, b,
        /*c_mod=*/(short)0, c, /*reuse_a=*/false, /*reuse_b=*/false);

    // D layout: VGPR r, lane L -> (M = r + 8*(L>=16), N = L&15)
    const int rbase = t0 + ((lane >> 4) << 3);
    #pragma unroll
    for (int r = 0; r < 8; ++r)
      P0[(size_t)(rbase + r) * GATES + col] = d[r];
  }
}

// ---------------------------------------------------------------------------
// Phase B: the sequential 3-layer recurrence. One block, 192 threads (6 waves).
// Transposed weights resident in LDS (~183 KB of the WGP's 320 KB).
// LDS float offsets:
// ---------------------------------------------------------------------------
#define OFF_WT0 0                       // [48][192]  W_hh0^T
#define OFF_WT1 (48 * 192)              // [96][192]  [W_ih1;W_hh1]^T
#define OFF_WT2 (OFF_WT1 + 96 * 192)    // [96][192]  [W_ih2;W_hh2]^T
#define OFF_B1  (OFF_WT2 + 96 * 192)    // [192] b_ih1+b_hh1
#define OFF_B2  (OFF_B1 + 192)          // [192] b_ih2+b_hh2
#define OFF_G   (OFF_B2 + 192)          // [192] activated gates, current layer
#define OFF_H   (OFF_G + 192)           // [144] h0|h1|h2 (16B aligned)
#define LDS_FLOATS (OFF_H + 160)

__device__ __forceinline__ float sigm(float v) {
  return 1.0f / (1.0f + __expf(-v));
}

__global__ __launch_bounds__(192) void lstm_seq(
    const float* __restrict__ P0,
    const float* __restrict__ Whh0,
    const float* __restrict__ Wih1, const float* __restrict__ Whh1,
    const float* __restrict__ bih1, const float* __restrict__ bhh1,
    const float* __restrict__ Wih2, const float* __restrict__ Whh2,
    const float* __restrict__ bih2, const float* __restrict__ bhh2,
    const float* __restrict__ prelu_a,
    const float* __restrict__ l1W, const float* __restrict__ l1b,
    const float* __restrict__ l2W, const float* __restrict__ l2b,
    float* __restrict__ out)
{
  extern __shared__ float lds[];
  const int j = threadIdx.x;                 // gate row 0..191

  // ---- one-time cooperative weight staging (transposed: k-major) ----
  for (int idx = j; idx < 48 * 192; idx += 192) {
    int k = idx / 192, jj = idx % 192;
    lds[OFF_WT0 + idx] = Whh0[jj * 48 + k];
  }
  for (int idx = j; idx < 96 * 192; idx += 192) {
    int k = idx / 192, jj = idx % 192;
    lds[OFF_WT1 + idx] = (k < 48) ? Wih1[jj * 48 + k] : Whh1[jj * 48 + (k - 48)];
    lds[OFF_WT2 + idx] = (k < 48) ? Wih2[jj * 48 + k] : Whh2[jj * 48 + (k - 48)];
  }
  lds[OFF_B1 + j] = bih1[j] + bhh1[j];
  lds[OFF_B2 + j] = bih2[j] + bhh2[j];
  if (j < 144) lds[OFF_H + j] = 0.0f;        // h0,h1,h2 = 0
  __syncthreads();

  const int gt = j / 48;                     // 0=i 1=f 2=g 3=o (PyTorch order)
  float c0 = 0.0f, c1 = 0.0f, c2 = 0.0f;     // cell state lives in threads 0..47

  float pin = P0[j];                         // projection for t=0
  #pragma unroll 1
  for (int t = 0; t < T_LEN; ++t) {
    if (t + 8 < T_LEN)                       // global_prefetch_b8, hide HBM latency
      __builtin_prefetch(&P0[(size_t)(t + 8) * GATES + j], 0, 1);
    const float pnext =
        (t + 1 < T_LEN) ? P0[(size_t)(t + 1) * GATES + j] : 0.0f;

    // ================= layer 0 =================
    {
      float hr[48];
      #pragma unroll
      for (int q = 0; q < 12; ++q) {
        float4 v = ((const float4*)(lds + OFF_H))[q];
        hr[4*q+0] = v.x; hr[4*q+1] = v.y; hr[4*q+2] = v.z; hr[4*q+3] = v.w;
      }
      float acc = pin;                       // W_ih0@x + both biases, precomputed
      #pragma unroll
      for (int k = 0; k < 48; ++k)
        acc = fmaf(lds[OFF_WT0 + k * 192 + j], hr[k], acc);
      lds[OFF_G + j] = (gt == 2) ? tanhf(acc) : sigm(acc);
    }
    __syncthreads();
    if (j < 48) {
      float ig = lds[OFF_G + j],       fg = lds[OFF_G + 48 + j];
      float gg = lds[OFF_G + 96 + j],  og = lds[OFF_G + 144 + j];
      c0 = fmaf(fg, c0, ig * gg);
      lds[OFF_H + j] = og * tanhf(c0);
    }
    __syncthreads();

    // ================= layer 1 ================= (input = h0_new, h1_old)
    {
      float hr[96];
      #pragma unroll
      for (int q = 0; q < 24; ++q) {
        float4 v = ((const float4*)(lds + OFF_H))[q];
        hr[4*q+0] = v.x; hr[4*q+1] = v.y; hr[4*q+2] = v.z; hr[4*q+3] = v.w;
      }
      float acc = lds[OFF_B1 + j];
      #pragma unroll
      for (int k = 0; k < 96; ++k)
        acc = fmaf(lds[OFF_WT1 + k * 192 + j], hr[k], acc);
      lds[OFF_G + j] = (gt == 2) ? tanhf(acc) : sigm(acc);
    }
    __syncthreads();
    if (j < 48) {
      float ig = lds[OFF_G + j],       fg = lds[OFF_G + 48 + j];
      float gg = lds[OFF_G + 96 + j],  og = lds[OFF_G + 144 + j];
      c1 = fmaf(fg, c1, ig * gg);
      lds[OFF_H + 48 + j] = og * tanhf(c1);
    }
    __syncthreads();

    // ================= layer 2 ================= (input = h1_new, h2_old)
    {
      float hr[96];
      #pragma unroll
      for (int q = 0; q < 24; ++q) {
        float4 v = ((const float4*)(lds + OFF_H + 48))[q];
        hr[4*q+0] = v.x; hr[4*q+1] = v.y; hr[4*q+2] = v.z; hr[4*q+3] = v.w;
      }
      float acc = lds[OFF_B2 + j];
      #pragma unroll
      for (int k = 0; k < 96; ++k)
        acc = fmaf(lds[OFF_WT2 + k * 192 + j], hr[k], acc);
      lds[OFF_G + j] = (gt == 2) ? tanhf(acc) : sigm(acc);
    }
    __syncthreads();
    if (j < 48) {
      float ig = lds[OFF_G + j],       fg = lds[OFF_G + 48 + j];
      float gg = lds[OFF_G + 96 + j],  og = lds[OFF_G + 144 + j];
      c2 = fmaf(fg, c2, ig * gg);
      lds[OFF_H + 96 + j] = og * tanhf(c2);
    }
    __syncthreads();

    pin = pnext;
  }

  // ================= epilogue: PReLU -> lin1 -> lin2 -> tanh =================
  if (j == 0) {
    const float av = prelu_a[0];
    float v6[6];
    for (int l = 0; l < 3; ++l) {
      for (int r = 0; r < 2; ++r) {
        float s = l1b[r];
        for (int u = 0; u < 48; ++u) {
          float hv = lds[OFF_H + l * 48 + u];
          hv = (hv > 0.0f) ? hv : av * hv;       // PReLU
          s = fmaf(l1W[r * 48 + u], hv, s);
        }
        v6[l * 2 + r] = s;
      }
    }
    for (int s2 = 0; s2 < 2; ++s2) {
      float s = l2b[s2];
      for (int m = 0; m < 6; ++m) s = fmaf(l2W[s2 * 6 + m], v6[m], s);
      out[s2] = tanhf(s);
    }
  }
}

// ---------------------------------------------------------------------------
extern "C" void kernel_launch(void* const* d_in, const int* in_sizes, int n_in,
                              void* d_out, int out_size, void* d_ws, size_t ws_size,
                              hipStream_t stream) {
  (void)in_sizes; (void)n_in; (void)out_size; (void)ws_size;
  const float* x    = (const float*)d_in[0];
  const float* Wih0 = (const float*)d_in[1];
  const float* Whh0 = (const float*)d_in[2];
  const float* bih0 = (const float*)d_in[3];
  const float* bhh0 = (const float*)d_in[4];
  const float* Wih1 = (const float*)d_in[5];
  const float* Whh1 = (const float*)d_in[6];
  const float* bih1 = (const float*)d_in[7];
  const float* bhh1 = (const float*)d_in[8];
  const float* Wih2 = (const float*)d_in[9];
  const float* Whh2 = (const float*)d_in[10];
  const float* bih2 = (const float*)d_in[11];
  const float* bhh2 = (const float*)d_in[12];
  const float* pa   = (const float*)d_in[13];
  const float* l1W  = (const float*)d_in[14];
  const float* l1b  = (const float*)d_in[15];
  const float* l2W  = (const float*)d_in[16];
  const float* l2b  = (const float*)d_in[17];
  float* out = (float*)d_out;
  float* P0  = (float*)d_ws;                // T*192 f32 = 192 MiB scratch

  // allow >64KB dynamic LDS (CDNA5 WGP has 320KB); ignore errors
  (void)hipFuncSetAttribute((const void*)lstm_seq,
                            hipFuncAttributeMaxDynamicSharedMemorySize,
                            LDS_FLOATS * 4);

  // Phase A: 2048 blocks x 8 waves = 16384 M-tiles (exact, no tail)
  lstm_inproj_wmma<<<MTILES / 8, 256, 0, stream>>>(x, Wih0, bih0, bhh0, P0);
  // Phase B: single workgroup sequential recurrence
  lstm_seq<<<1, 192, LDS_FLOATS * 4, stream>>>(P0, Whh0,
      Wih1, Whh1, bih1, bhh1, Wih2, Whh2, bih2, bhh2,
      pa, l1W, l1b, l2W, l2b, out);
}
